// BertEncoderCTC_4234837754337
// MI455X (gfx1250) — matrix operand order repaired and checked by reading, not compiled, not run on hardware
//
#include <hip/hip_runtime.h>
#include <math.h>
#include <stdint.h>

// ---------------------------------------------------------------------------
// Types / helpers
// ---------------------------------------------------------------------------
typedef __attribute__((ext_vector_type(16))) __bf16 bf16x16;
typedef __attribute__((ext_vector_type(8)))  __bf16 bf16x8;
typedef __attribute__((ext_vector_type(8)))  float  f32x8;

#define DEV static __device__ __forceinline__

DEV __bf16 f2bf(float f){
  unsigned u = __builtin_bit_cast(unsigned, f);
  unsigned r = u + 0x7FFFu + ((u >> 16) & 1u);
  unsigned short h = (unsigned short)(r >> 16);
  return __builtin_bit_cast(__bf16, h);
}

// async 16B global -> LDS copy (per lane), tracked by ASYNCcnt
DEV void async_cp16(unsigned lds_off, const void* gaddr){
  asm volatile("global_load_async_to_lds_b128 %0, %1, off"
               :: "v"(lds_off), "v"((unsigned long long)(uintptr_t)gaddr)
               : "memory");
}
DEV void async_wait0(){
  asm volatile("s_wait_asynccnt 0x0" ::: "memory");
}
DEV unsigned lds_off32(const void* p){ return (unsigned)(uintptr_t)p; }

// A fragment (16x32 MxK, bf16): lane&15 = row M; kg=lane>>4 selects
// K {kg*8..+7} and {16+kg*8..+7}  (ISA 7.12.2 layout)
DEV bf16x16 load_fragA(const __bf16* row, int kg){
  bf16x8 lo = *(const bf16x8*)(row + kg*8);
  bf16x8 hi = *(const bf16x8*)(row + 16 + kg*8);
  bf16x16 r;
  #pragma unroll
  for (int i=0;i<8;++i){ r[i]=lo[i]; r[i+8]=hi[i]; }
  return r;
}
// B fragment (32x16 KxN): lane&15 = column N, kg selects
// K {kg*8..+7} and {16+kg*8..+7} (same as A)
DEV bf16x16 load_fragB(const __bf16* col, int kg){
  bf16x8 lo = *(const bf16x8*)(col + kg*8);
  bf16x8 hi = *(const bf16x8*)(col + 16 + kg*8);
  bf16x16 r;
  #pragma unroll
  for (int i=0;i<8;++i){ r[i]=lo[i]; r[i+8]=hi[i]; }
  return r;
}

DEV f32x8 wmma_bf16(bf16x16 a, bf16x16 b, f32x8 c){
  return __builtin_amdgcn_wmma_f32_16x16x32_bf16(false, a, false, b, (short)0, c,
                                                 false, false);
}

DEV bf16x8 zero_bf8(){
  __bf16 z = f2bf(0.f);
  bf16x8 r = {z,z,z,z,z,z,z,z};
  return r;
}

// ---------------------------------------------------------------------------
// Batched bf16 GEMM:  C[m,n] = sum_k A[m,k]*B[n,k] + bias[n]
// A: (M,K) bf16 row-major; B: (N,K) bf16 row-major. Double-buffered LDS,
// staged with GLOBAL_LOAD_ASYNC_TO_LDS_B128 overlapped with WMMA.
// Outputs: Cf (f32, optional) and/or Cb (bf16, optional); gelu optional.
// Batch z offset: (z/innerH)*sXo + (z%innerH)*sXi   (element units)
// Requires K%32==0, N%64==0 (true for all calls); M guarded (zero-filled rows).
// ---------------------------------------------------------------------------
__global__ __launch_bounds__(256)
void gemm_bf16_kernel(const __bf16* __restrict__ A, int lda, long long sAo, long long sAi,
                      const __bf16* __restrict__ Bm, int ldb, long long sBo, long long sBi,
                      const float* __restrict__ bias,
                      float* __restrict__ Cf, __bf16* __restrict__ Cb,
                      int ldc, long long sCo, long long sCi,
                      int M, int N, int K, int gelu, int innerH)
{
  __shared__ alignas(16) __bf16 As[2][64][40];
  __shared__ alignas(16) __bf16 Bs[2][64][40];
  const int tid = threadIdx.x, lane = tid & 31, wave = tid >> 5;
  const int wi = wave & 1, wj = wave >> 1;
  const int l16 = lane & 15, kg = lane >> 4;
  const int bm = blockIdx.y << 6, bn = blockIdx.x << 6;

  long long z = blockIdx.z, zo = z / innerH, zi = z - zo*innerH;
  A  += zo*sAo + zi*sAi;
  Bm += zo*sBo + zi*sBi;
  long long coff = zo*sCo + zi*sCi;

  const int am = tid >> 2, akk = (tid & 3) << 3;  // 64 rows x 4 lanes x 8 bf16
  const int gm = bm + am;
  const bool mok = gm < M;
  const __bf16* gA = A + (long long)gm*lda + akk;
  const __bf16* gB = Bm + (long long)(bn+am)*ldb + akk;
  const unsigned ldsA0 = lds_off32(&As[0][am][akk]);
  const unsigned ldsA1 = lds_off32(&As[1][am][akk]);
  const unsigned ldsB0 = lds_off32(&Bs[0][am][akk]);
  const unsigned ldsB1 = lds_off32(&Bs[1][am][akk]);
  if (!mok){
    bf16x8 zb = zero_bf8();
    *(bf16x8*)&As[0][am][akk] = zb;
    *(bf16x8*)&As[1][am][akk] = zb;
  }

  // prologue: stage K-step 0 into buffer 0
  if (mok) async_cp16(ldsA0, gA);
  async_cp16(ldsB0, gB);
  async_wait0();
  __syncthreads();

  const f32x8 zero8 = {0,0,0,0,0,0,0,0};
  f32x8 acc0 = zero8, acc1 = zero8;
  const int ksteps = K >> 5;

  for (int s = 0; s < ksteps; ++s){
    const int cur = s & 1;
    if (s + 1 < ksteps){          // prefetch next tile into other buffer
      const int nk = (s + 1) << 5;
      if (mok) async_cp16(cur ? ldsA0 : ldsA1, gA + nk);
      async_cp16(cur ? ldsB0 : ldsB1, gB + nk);
    }
    const __bf16* as = &As[cur][0][0];
    const __bf16* bs = &Bs[cur][0][0];
    bf16x16 bf = load_fragB(bs + ((wj<<4) + l16)*40, kg);
    bf16x16 a0 = load_fragA(as + ((wi<<5) + l16)*40, kg);
    bf16x16 a1 = load_fragA(as + ((wi<<5) + 16 + l16)*40, kg);
    acc0 = wmma_bf16(a0, bf, acc0);
    acc1 = wmma_bf16(a1, bf, acc1);
    async_wait0();
    __syncthreads();
  }

  const int n  = bn + (wj<<4) + l16;
  const float bb = bias ? bias[n] : 0.f;
  const int mb = bm + (wi<<5) + (kg<<3);
  #pragma unroll
  for (int r=0;r<8;++r){
    float v0 = acc0[r] + bb;
    float v1 = acc1[r] + bb;
    if (gelu){
      v0 = 0.5f*v0*(1.f + erff(v0*0.70710678118654752f));
      v1 = 0.5f*v1*(1.f + erff(v1*0.70710678118654752f));
    }
    int m0 = mb + r, m1 = mb + 16 + r;
    if (m0 < M){
      if (Cf) Cf[coff + (long long)m0*ldc + n] = v0;
      if (Cb) Cb[coff + (long long)m0*ldc + n] = f2bf(v0);
    }
    if (m1 < M){
      if (Cf) Cf[coff + (long long)m1*ldc + n] = v1;
      if (Cb) Cb[coff + (long long)m1*ldc + n] = f2bf(v1);
    }
  }
}

// ---------------------------------------------------------------------------
// Fused attention-scores kernel (bf16 inputs, async-staged):
//   ac = (q+u).k^T (64x64, K=64) ; bd = (q+v).p_band^T (64x128, K=64)
//   scores = (ac + bd[ki-qi+63]) * 0.125, masked      (rel_shift identity)
// ---------------------------------------------------------------------------
__global__ __launch_bounds__(256)
void scores_kernel(const __bf16* __restrict__ qu, const __bf16* __restrict__ qv,
                   const __bf16* __restrict__ kb, const __bf16* __restrict__ pb,
                   const unsigned char* __restrict__ mask,
                   float* __restrict__ sc)
{
  const int Nn=512, Dd=768, Hh=12, DKk=64, Pp=1023;
  __shared__ alignas(16) __bf16 Qs[64][40];
  __shared__ alignas(16) __bf16 Ks[64][40];
  __shared__ alignas(16) __bf16 Ps[128][40];
  __shared__ float acL[64][72];
  __shared__ float bdL[64][136];

  const int tid  = threadIdx.x, lane = tid & 31, wave = tid >> 5;
  const int wi = wave & 1, wj = wave >> 1;
  const int l16 = lane & 15, kg = lane >> 4;
  const int k0 = blockIdx.x << 6, q0 = blockIdx.y << 6;
  const int b = blockIdx.z / Hh, hd = (blockIdx.z % Hh) * DKk;

  const __bf16* Aq = qu + ((long long)b*Nn + q0)*Dd + hd;
  const __bf16* Av = qv + ((long long)b*Nn + q0)*Dd + hd;
  const __bf16* Bk = kb + ((long long)b*Nn + k0)*Dd + hd;

  const int am = tid >> 2, akk = (tid & 3) << 3;
  const unsigned ldsQ = lds_off32(&Qs[am][akk]);
  const unsigned ldsK = lds_off32(&Ks[am][akk]);
  const __bf16* gQ = Aq + (long long)am*Dd + akk;
  const __bf16* gK = Bk + (long long)am*Dd + akk;

  // Ps rows owned by this thread (2 of 128); zero out-of-band rows once
  const int jbase = k0 - q0 + Nn - 64;
  const int j0 = tid >> 2;
  const int jg0 = jbase + j0, jg1 = jbase + j0 + 64;
  const bool p0ok = (jg0 >= 0) && (jg0 < Pp);
  const bool p1ok = (jg1 >= 0) && (jg1 < Pp);
  const unsigned ldsP0 = lds_off32(&Ps[j0][akk]);
  const unsigned ldsP1 = lds_off32(&Ps[j0+64][akk]);
  const __bf16* gP0 = pb + (long long)jg0*Dd + hd + akk;
  const __bf16* gP1 = pb + (long long)jg1*Dd + hd + akk;
  {
    bf16x8 zb = zero_bf8();
    if (!p0ok) *(bf16x8*)&Ps[j0][akk]    = zb;
    if (!p1ok) *(bf16x8*)&Ps[j0+64][akk] = zb;
  }

  const f32x8 zero8 = {0,0,0,0,0,0,0,0};

  // ---- phase 1: ac tile ----
  f32x8 acc0 = zero8, acc1 = zero8;
  for (int kk0 = 0; kk0 < DKk; kk0 += 32){
    async_cp16(ldsQ, gQ + kk0);
    async_cp16(ldsK, gK + kk0);
    async_wait0();
    __syncthreads();
    bf16x16 bf = load_fragB(&Ks[(wj<<4) + l16][0], kg);
    bf16x16 a0 = load_fragA(&Qs[(wi<<5) + l16][0], kg);
    bf16x16 a1 = load_fragA(&Qs[(wi<<5) + 16 + l16][0], kg);
    acc0 = wmma_bf16(a0, bf, acc0);
    acc1 = wmma_bf16(a1, bf, acc1);
    __syncthreads();
  }
  {
    int n = (wj<<4) + l16;
    #pragma unroll
    for (int r=0;r<8;++r){
      acL[(wi<<5) + (kg<<3) + r][n]      = acc0[r];
      acL[(wi<<5) + 16 + (kg<<3) + r][n] = acc1[r];
    }
  }

  // ---- phase 2: banded bd tile (64 x 128) ----
  f32x8 bd00 = zero8, bd01 = zero8, bd10 = zero8, bd11 = zero8;
  const __bf16* gV = Av + (long long)am*Dd + akk;
  for (int kk0 = 0; kk0 < DKk; kk0 += 32){
    async_cp16(ldsQ, gV + kk0);
    if (p0ok) async_cp16(ldsP0, gP0 + kk0);
    if (p1ok) async_cp16(ldsP1, gP1 + kk0);
    async_wait0();
    __syncthreads();
    bf16x16 a0 = load_fragA(&Qs[(wi<<5) + l16][0], kg);
    bf16x16 a1 = load_fragA(&Qs[(wi<<5) + 16 + l16][0], kg);
    bf16x16 p0 = load_fragB(&Ps[(wj<<5) + l16][0], kg);
    bf16x16 p1 = load_fragB(&Ps[(wj<<5) + 16 + l16][0], kg);
    bd00 = wmma_bf16(a0, p0, bd00);
    bd01 = wmma_bf16(a0, p1, bd01);
    bd10 = wmma_bf16(a1, p0, bd10);
    bd11 = wmma_bf16(a1, p1, bd11);
    __syncthreads();
  }
  {
    #pragma unroll
    for (int r=0;r<8;++r){
      bdL[(wi<<5) + (kg<<3) + r][(wj<<5) + l16]           = bd00[r];
      bdL[(wi<<5) + (kg<<3) + r][(wj<<5) + 16 + l16]      = bd01[r];
      bdL[(wi<<5) + 16 + (kg<<3) + r][(wj<<5) + l16]      = bd10[r];
      bdL[(wi<<5) + 16 + (kg<<3) + r][(wj<<5) + 16 + l16] = bd11[r];
    }
  }
  __syncthreads();

  // ---- phase 3: combine with shift, scale, mask, store ----
  {
    int qi = tid >> 2, kb0 = (tid & 3) << 4;
    float* out = sc + ((long long)blockIdx.z*Nn + (q0+qi))*Nn + k0;
    const unsigned char* mr = mask + ((long long)b*Nn + (q0+qi))*Nn + k0;
    #pragma unroll
    for (int i=0;i<16;++i){
      int ki = kb0 + i;
      float v = (acL[qi][ki] + bdL[qi][ki - qi + 63]) * 0.125f;
      if (mr[ki]) v = -10000.f;
      out[ki] = v;
    }
  }
}

// ---------------------------------------------------------------------------
// Row softmax over N=512; writes bf16 attention (may alias the f32 scores:
// each block fully reads its row before any write).
// ---------------------------------------------------------------------------
__global__ __launch_bounds__(256)
void softmax_kernel(const float* __restrict__ sc, const unsigned char* __restrict__ mask,
                    __bf16* __restrict__ attnb)
{
  const int Nn=512, Hh=12;
  __shared__ float red[256];
  int row = blockIdx.x;                 // over B*H*N
  int q = row % Nn;
  int b = row / (Hh*Nn);
  const float* s = sc + (long long)row*Nn;
  const unsigned char* mr = mask + ((long long)b*Nn + q)*Nn;
  int t = threadIdx.x;

  float v0 = s[t], v1 = s[t+256];
  red[t] = fmaxf(v0, v1); __syncthreads();
  for (int off=128; off; off>>=1){ if (t<off) red[t]=fmaxf(red[t],red[t+off]); __syncthreads(); }
  float mx = red[0]; __syncthreads();

  float e0 = __expf(v0-mx), e1 = __expf(v1-mx);
  red[t] = e0+e1; __syncthreads();
  for (int off=128; off; off>>=1){ if (t<off) red[t]+=red[t+off]; __syncthreads(); }
  float inv = 1.f / red[0];
  __syncthreads();   // all reads of this row done before aliased writes

  __bf16* o = attnb + (long long)row*Nn;
  o[t]     = f2bf(mr[t]     ? 0.f : e0*inv);
  o[t+256] = f2bf(mr[t+256] ? 0.f : e1*inv);
}

// ---------------------------------------------------------------------------
// out = LayerNorm(a + b)*g + beta  (row length 768); optional bf16 copy
// ---------------------------------------------------------------------------
__global__ __launch_bounds__(256)
void addln_kernel(const float* __restrict__ A, const float* __restrict__ Bb,
                  const float* __restrict__ g, const float* __restrict__ beta,
                  float* __restrict__ outF, __bf16* __restrict__ outB)
{
  __shared__ float red[256];
  int row = blockIdx.x, t = threadIdx.x;
  const float* pa = A  + (long long)row*768;
  const float* pb = Bb + (long long)row*768;
  float x0 = pa[t]     + pb[t];
  float x1 = pa[t+256] + pb[t+256];
  float x2 = pa[t+512] + pb[t+512];
  red[t] = x0+x1+x2; __syncthreads();
  for (int off=128; off; off>>=1){ if (t<off) red[t]+=red[t+off]; __syncthreads(); }
  float mean = red[0]*(1.f/768.f); __syncthreads();
  float d0=x0-mean, d1=x1-mean, d2=x2-mean;
  red[t] = d0*d0+d1*d1+d2*d2; __syncthreads();
  for (int off=128; off; off>>=1){ if (t<off) red[t]+=red[t+off]; __syncthreads(); }
  float inv = rsqrtf(red[0]*(1.f/768.f) + 1e-5f);
  float y0 = d0*inv*g[t]     + beta[t];
  float y1 = d1*inv*g[t+256] + beta[t+256];
  float y2 = d2*inv*g[t+512] + beta[t+512];
  if (outF){
    float* po = outF + (long long)row*768;
    po[t] = y0; po[t+256] = y1; po[t+512] = y2;
  }
  if (outB){
    __bf16* po = outB + (long long)row*768;
    po[t] = f2bf(y0); po[t+256] = f2bf(y1); po[t+512] = f2bf(y2);
  }
}

// cat(bf16) = [x (B,N,128), residual (B,N,768)] -> (4096, 896)
__global__ void concat_kernel(const float* __restrict__ x, const float* __restrict__ res,
                              __bf16* __restrict__ out)
{
  long long i = (long long)blockIdx.x*256 + threadIdx.x;
  int m = (int)(i / 896), c = (int)(i % 896);
  float v = (c < 128) ? x[(long long)m*128 + c] : res[(long long)m*768 + (c-128)];
  out[i] = f2bf(v);
}

// f32 -> bf16 convert
__global__ void cvt_kernel(const float* __restrict__ in, __bf16* __restrict__ out, int n)
{
  int i = blockIdx.x*256 + threadIdx.x;
  if (i < n) out[i] = f2bf(in[i]);
}

// vT[b,h,d,k] = v[b,k,h*64+d]   (bf16 -> bf16), output (B,H,64,512)
__global__ void vtrans_kernel(const __bf16* __restrict__ v, __bf16* __restrict__ vt)
{
  long long i = (long long)blockIdx.x*256 + threadIdx.x;   // 8*12*64*512
  int k  = (int)(i & 511);
  long long t = i >> 9;
  int d  = (int)(t & 63);
  long long t2 = t >> 6;
  int h  = (int)(t2 % 12);
  int b  = (int)(t2 / 12);
  vt[i] = v[((long long)b*512 + k)*768 + h*64 + d];
}

// bias_u = bq + pos_bias_u.flat ; bias_v = bq + pos_bias_v.flat
__global__ void biasuv_kernel(const float* __restrict__ bq,
                              const float* __restrict__ pbu, const float* __restrict__ pbv,
                              float* __restrict__ bu, float* __restrict__ bv)
{
  int i = blockIdx.x*256 + threadIdx.x;
  if (i < 768)       bu[i] = bq[i] + pbu[i];
  else if (i < 1536) bv[i-768] = bq[i-768] + pbv[i-768];
}

// ---------------------------------------------------------------------------
extern "C" void kernel_launch(void* const* d_in, const int* in_sizes, int n_in,
                              void* d_out, int out_size, void* d_ws, size_t ws_size,
                              hipStream_t stream)
{
  const int Bn=8, Nn=512, Hh=12, DKk=64, Aa=512, FFn=3072, Pp=1023;
  const int BN = Bn*Nn; // 4096

  const float* x        = (const float*)d_in[0];
  const float* pos_emb  = (const float*)d_in[2];
  const float* residual = (const float*)d_in[3];
  const unsigned char* mask = (const unsigned char*)d_in[4];
  const float* fuse_W   = (const float*)d_in[5];
  const float* fuse_b   = (const float*)d_in[6];
  const float* pp_W     = (const float*)d_in[7];
  const float* pp_b     = (const float*)d_in[8];
  const float* Wq       = (const float*)d_in[9];
  const float* bq       = (const float*)d_in[10];
  const float* Wk       = (const float*)d_in[11];
  const float* bk       = (const float*)d_in[12];
  const float* Wv       = (const float*)d_in[13];
  const float* bvv      = (const float*)d_in[14];
  const float* Wout     = (const float*)d_in[15];
  const float* bout     = (const float*)d_in[16];
  const float* Wpos     = (const float*)d_in[17];
  const float* pbu      = (const float*)d_in[18];
  const float* pbv      = (const float*)d_in[19];
  const float* ln1g     = (const float*)d_in[20];
  const float* ln1b     = (const float*)d_in[21];
  const float* Wi       = (const float*)d_in[22];
  const float* bi       = (const float*)d_in[23];
  const float* Wo       = (const float*)d_in[24];
  const float* bo       = (const float*)d_in[25];
  const float* ln2g     = (const float*)d_in[26];
  const float* ln2b     = (const float*)d_in[27];
  const float* Wac      = (const float*)d_in[28];
  const float* bac      = (const float*)d_in[29];

  // ---- workspace layout (byte bump allocator, 256B aligned blocks) ----
  char* base = (char*)d_ws;
  size_t off = 0;
  auto alloc = [&](size_t bytes)->void*{
    void* p = base + off; off += (bytes + 255) & ~(size_t)255; return p;
  };
  // bf16 region shared by cat (dies early) and ff1 (late)
  __bf16* catb   = (__bf16*)alloc((size_t)12582912*2);
  __bf16* ff1b   = catb;
  __bf16* fuseWb = (__bf16*)alloc((size_t)688128*2);
  __bf16* ppWb   = (__bf16*)alloc((size_t)393216*2);
  __bf16* Wqb    = (__bf16*)alloc((size_t)589824*2);
  __bf16* Wkb    = (__bf16*)alloc((size_t)589824*2);
  __bf16* Wvb    = (__bf16*)alloc((size_t)589824*2);
  __bf16* Wposb  = (__bf16*)alloc((size_t)589824*2);
  __bf16* Woutb  = (__bf16*)alloc((size_t)589824*2);
  __bf16* Wib    = (__bf16*)alloc((size_t)2359296*2);
  __bf16* Wob    = (__bf16*)alloc((size_t)2359296*2);
  __bf16* Wacb   = (__bf16*)alloc((size_t)393216*2);
  __bf16* peW    = (__bf16*)alloc((size_t)523776*2);   // pos_emb bf16
  __bf16* hb     = (__bf16*)alloc((size_t)3145728*2);
  float*  hf     = (float*)alloc((size_t)3145728*4);
  __bf16* peb    = (__bf16*)alloc((size_t)785664*2);
  __bf16* qub    = (__bf16*)alloc((size_t)3145728*2);
  __bf16* qvb    = (__bf16*)alloc((size_t)3145728*2);
  __bf16* kbb    = (__bf16*)alloc((size_t)3145728*2);
  __bf16* vbb    = (__bf16*)alloc((size_t)3145728*2);
  __bf16* pbb    = (__bf16*)alloc((size_t)785664*2);
  __bf16* vT     = (__bf16*)alloc((size_t)3145728*2);
  float*  sc     = (float*)alloc((size_t)25165824*4);
  __bf16* ctxb   = (__bf16*)alloc((size_t)3145728*2);
  float*  bu     = (float*)alloc(768*4);
  float*  bv2    = (float*)alloc(768*4);
  // lifetime aliases
  __bf16* attnb = (__bf16*)sc;                  // softmax output (in place)
  float*  aoutf = (float*)qub;                  // qub+qvb dead after scores
  float*  h1f   = (float*)kbb;                  // kbb+vbb dead after scores
  __bf16* h1b   = vT;                           // vT dead after ctx gemm
  float*  ffof  = sc;                           // attn dead after ctx gemm
  __bf16* h2b   = (__bf16*)(sc + 8388608);      // 32MB into sc region

  float* out_ac = (float*)d_out;                // B*N*A
  float* h2     = out_ac + (long long)Bn*Nn*Aa;

  auto gemm = [&](const __bf16* A_, int lda, const __bf16* B_, int ldb,
                  const float* bias_, float* Cf_, __bf16* Cb_, int ldc,
                  int M, int N, int K, int gelu,
                  int batch=1, int innerH=1,
                  long long sAo=0, long long sAi=0, long long sBo=0, long long sBi=0,
                  long long sCo=0, long long sCi=0){
    dim3 g((N+63)/64, (M+63)/64, batch);
    gemm_bf16_kernel<<<g, 256, 0, stream>>>(A_, lda, sAo, sAi, B_, ldb, sBo, sBi,
                                            bias_, Cf_, Cb_, ldc, sCo, sCi,
                                            M, N, K, gelu, innerH);
  };
  auto cvt = [&](const float* in, __bf16* outp, int n){
    cvt_kernel<<<(n+255)/256, 256, 0, stream>>>(in, outp, n);
  };

  // 0) one-time f32->bf16 weight conversion
  cvt(fuse_W, fuseWb, 688128);
  cvt(pp_W,   ppWb,   393216);
  cvt(Wq,     Wqb,    589824);
  cvt(Wk,     Wkb,    589824);
  cvt(Wv,     Wvb,    589824);
  cvt(Wpos,   Wposb,  589824);
  cvt(Wout,   Woutb,  589824);
  cvt(Wi,     Wib,    2359296);
  cvt(Wo,     Wob,    2359296);
  cvt(Wac,    Wacb,   393216);
  cvt(pos_emb, peW,   523776);

  // 1) cat = [x, residual] (bf16)
  concat_kernel<<<(BN*896)/256, 256, 0, stream>>>(x, residual, catb);
  // 2) folded biases
  biasuv_kernel<<<6, 256, 0, stream>>>(bq, pbu, pbv, bu, bv2);
  // 3) h = cat @ fuse_W^T + fuse_b  (f32 + bf16)
  gemm(catb, 896, fuseWb, 896, fuse_b, hf, hb, 768, BN, 768, 896, 0);
  // 4) pe = pos_emb @ posproj_W^T + posproj_b (bf16)
  gemm(peW, Aa, ppWb, Aa, pp_b, nullptr, peb, 768, Pp, 768, Aa, 0);
  // 5-8) qu, qv, k, v (bf16)
  gemm(hb, 768, Wqb, 768, bu,  nullptr, qub, 768, BN, 768, 768, 0);
  gemm(hb, 768, Wqb, 768, bv2, nullptr, qvb, 768, BN, 768, 768, 0);
  gemm(hb, 768, Wkb, 768, bk,  nullptr, kbb, 768, BN, 768, 768, 0);
  gemm(hb, 768, Wvb, 768, bvv, nullptr, vbb, 768, BN, 768, 768, 0);
  // 9) p = pe @ Wpos^T (bf16)
  gemm(peb, 768, Wposb, 768, nullptr, nullptr, pbb, 768, Pp, 768, 768, 0);
  // 10) v transpose -> (B,H,DK,N)
  vtrans_kernel<<<3145728/256, 256, 0, stream>>>(vbb, vT);
  // 11) fused ac + shifted bd -> scores (f32)
  scores_kernel<<<dim3(Nn/64, Nn/64, Bn*Hh), 256, 0, stream>>>(qub, qvb, kbb, pbb, mask, sc);
  // 12) softmax -> attn (bf16, in place over sc)
  softmax_kernel<<<Bn*Hh*Nn, 256, 0, stream>>>(sc, mask, attnb);
  // 13) ctx[b,:,h*DK:] = attn[b,h] @ vT[b,h]^T  (batched per (b,h))
  gemm(attnb, Nn, vT, Nn, nullptr, nullptr, ctxb, 768, Nn, DKk, Nn, 0,
       Bn*Hh, Hh,
       (long long)Hh*Nn*Nn, (long long)Nn*Nn,
       (long long)Hh*DKk*Nn, (long long)DKk*Nn,
       (long long)Nn*768, (long long)DKk);
  // 14) attn_out = ctx @ Wout^T + bout (f32)
  gemm(ctxb, 768, Woutb, 768, bout, aoutf, nullptr, 768, BN, 768, 768, 0);
  // 15) h1 = LN(attn_out + h) (f32 + bf16)
  addln_kernel<<<BN, 256, 0, stream>>>(aoutf, hf, ln1g, ln1b, h1f, h1b);
  // 16) ff1 = gelu(h1 @ Wi^T + bi) (bf16)
  gemm(h1b, 768, Wib, 768, bi, nullptr, ff1b, FFn, BN, FFn, 768, 1);
  // 17) ffo = ff1 @ Wo^T + bo (f32)
  gemm(ff1b, FFn, Wob, FFn, bo, ffof, nullptr, 768, BN, 768, FFn, 0);
  // 18) h2 = LN(ffo + h1) -> output 2 (+ bf16 copy)
  addln_kernel<<<BN, 256, 0, stream>>>(ffof, h1f, ln2g, ln2b, h2, h2b);
  // 19) out_ac = h2 @ Wac^T + bac -> output 1
  gemm(h2b, 768, Wacb, 768, bac, out_ac, nullptr, Aa, BN, Aa, 768, 0);

  (void)in_sizes; (void)n_in; (void)out_size; (void)ws_size;
}
